// Embedding_68719477131
// MI455X (gfx1250) — matrix operand order repaired and checked by reading, not compiled, not run on hardware
//
#include <hip/hip_runtime.h>

// ---------------------------------------------------------------------------
// CDNA5 / gfx1250 point-embedding pipeline (wave32, WMMA f16->f32).
// ~21 GFLOP total, latency-bound -> 7 fused kernels, LDS-resident tiles.
// All WMMA fragments are fed by exactly two 128-bit LDS/global loads:
//   * activations stored position-major [pos][chan], rows padded to C+8 halfs
//     (row strides 20/36/68/196 dwords -> 16 lanes hit 16 distinct banks)
//   * weights stored row-major [cout][cin], cin padded to cin+8
//   * D write-back packed as one uint4 (8 f16 channels) per lane row
// ---------------------------------------------------------------------------

typedef __attribute__((ext_vector_type(16))) _Float16 v16h;
typedef __attribute__((ext_vector_type(8)))  float    v8f;

union FragH  { v16h v; _Float16 h[16]; uint4 u[2]; };
union FragF  { v8f  v; float     f[8]; };
union Pack8  { _Float16 h[8];  uint4 u;    };
union Pack40 { _Float16 h[40]; uint4 u[5]; };
union Pack64 { _Float16 h[64]; uint4 u[8]; };
union PackF4 { float f[4];     float4 v;   };

__device__ __forceinline__ float lrelu(float v) { return fmaxf(v, 0.2f * v); }

// A fragment 16x32 f16 from row-major [m][k] (lda mult of 8, k0 mult of 32).
// ISA: lane m = lane&15; K octets {kb..kb+7, kb+16..kb+23}, kb = (lane>>4)*8.
__device__ __forceinline__ v16h frag_a_rm(const _Float16* a, int lda, int m0, int k0) {
  int lane = threadIdx.x & 31;
  const _Float16* p = a + (m0 + (lane & 15)) * lda + k0 + ((lane >> 4) << 3);
  FragH fr;
  fr.u[0] = *(const uint4*)p;
  fr.u[1] = *(const uint4*)(p + 16);
  return fr.v;
}

// B fragment 32x16 f16 from position-major [n][k] (ldb mult of 8, k0 mult of 32).
// ISA: lane n = lane&15; K octets {kb..kb+7, kb+16..kb+23}, kb = (lane>>4)*8.
__device__ __forceinline__ v16h frag_b_pm(const _Float16* b, int ldb, int k0, int n0) {
  int lane = threadIdx.x & 31;
  const _Float16* p = b + (n0 + (lane & 15)) * ldb + k0 + ((lane >> 4) << 3);
  FragH fr;
  fr.u[0] = *(const uint4*)p;
  fr.u[1] = *(const uint4*)(p + 16);
  return fr.v;
}

#define WMMA_F16(acc, af, bf) \
  acc = __builtin_amdgcn_wmma_f32_16x16x32_f16(false, af, false, bf, (short)0, acc, false, false)

// D fragment post-op + packed store to position-major f16 [n][oc], ld mult 8.
__device__ __forceinline__ void store_d_bn_lrelu(
    _Float16* dst, int ld, int n0, int mb, const v8f& acc,
    const float* s, const float* bia) {
  int lane = threadIdx.x & 31;
  int n  = n0 + (lane & 15);
  int m  = mb + ((lane >> 4) << 3);
  FragF fr; fr.v = acc;
  Pack8 pk;
#pragma unroll
  for (int r = 0; r < 8; ++r) {
    int oc = m + r;
    pk.h[r] = (_Float16)lrelu(fr.f[r] * s[oc] + bia[oc]);
  }
  *(uint4*)(dst + n * ld + m) = pk.u;
}

// ---------------------------------------------------------------------------
// 0) global min/max of x
// ---------------------------------------------------------------------------
__global__ __launch_bounds__(256) void k_minmax(const float* x, int n, float* mnmx) {
  __shared__ float smn[256], smx[256];
  float mn = 3.4e38f, mx = -3.4e38f;
  for (int i = threadIdx.x; i < n; i += 256) {
    float v = x[i];
    mn = fminf(mn, v); mx = fmaxf(mx, v);
  }
  smn[threadIdx.x] = mn; smx[threadIdx.x] = mx;
  __syncthreads();
  for (int s = 128; s > 0; s >>= 1) {
    if (threadIdx.x < (unsigned)s) {
      smn[threadIdx.x] = fminf(smn[threadIdx.x], smn[threadIdx.x + s]);
      smx[threadIdx.x] = fmaxf(smx[threadIdx.x], smx[threadIdx.x + s]);
    }
    __syncthreads();
  }
  if (threadIdx.x == 0) { mnmx[0] = smn[0]; mnmx[1] = smx[0]; }
}

// ---------------------------------------------------------------------------
// 1) weight f32 -> f16 with padded leading dimension (zeros in pad region)
// ---------------------------------------------------------------------------
__global__ void k_cvt_pad(const float* src, _Float16* dst, int O, int I, int LD) {
  int i = blockIdx.x * blockDim.x + threadIdx.x;
  if (i < O * LD) {
    int oc = i / LD, c = i % LD;
    dst[i] = (_Float16)(c < I ? src[oc * I + c] : 0.f);
  }
}

// ---------------------------------------------------------------------------
// 2) KNN in 3-D: full cloud in LDS, per-thread top-32 insertion list.
//    neg_dist = 2*dot - sq_q - sq_c ; ties -> lower index first (lax.top_k).
// ---------------------------------------------------------------------------
__global__ __launch_bounds__(256) void k_knn1(const float* x, int* idx1, int B, int N) {
  extern __shared__ char smem[];
  float* px = (float*)smem;
  float* py = px + N;
  float* pz = py + N;
  float* sq = pz + N;
  int blk = blockIdx.x;
  int b = blk / (N / 256);
  int qbase = (blk % (N / 256)) * 256;
  const float* xb = x + (size_t)b * 3 * N;
  for (int i = threadIdx.x; i < N; i += 256) {
    float a = xb[i], bb = xb[N + i], c = xb[2 * N + i];
    px[i] = a; py[i] = bb; pz[i] = c;
    sq[i] = a * a + bb * bb + c * c;
  }
  __syncthreads();
  int q = qbase + threadIdx.x;
  float qx = px[q], qy = py[q], qz = pz[q], qs = sq[q];
  float bv[32]; int bi[32];
#pragma unroll
  for (int k = 0; k < 32; ++k) { bv[k] = -3.4e38f; bi[k] = 0x7fffffff; }
  for (int m = 0; m < N; ++m) {
    float v = 2.f * (qx * px[m] + qy * py[m] + qz * pz[m]) - qs - sq[m];
    if (v > bv[31] || (v == bv[31] && m < bi[31])) {
      int j = 31;
      while (j > 0 && (bv[j - 1] < v || (bv[j - 1] == v && bi[j - 1] > m))) {
        bv[j] = bv[j - 1]; bi[j] = bi[j - 1]; --j;
      }
      bv[j] = v; bi[j] = m;
    }
  }
  for (int k = 0; k < 32; ++k) idx1[((size_t)b * N + q) * 32 + k] = bi[k];
}

// ---------------------------------------------------------------------------
// 3) stage1: gather 3D [center,diff] (cin 6 pad 32) -> W1 -> W2 -> max over K
//    8 points (256 positions) per 8-wave workgroup.  Rows padded +8 halfs.
// ---------------------------------------------------------------------------
__global__ __launch_bounds__(256) void k_stage1(
    const float* x, const int* idx1,
    const _Float16* w1g, const _Float16* w2g,
    const float* s1, const float* b1, const float* s2, const float* b2,
    float* f1fT, _Float16* f1hT, int B, int N)
{
  const int POS = 256, PTS = 8;
  const int LDI = 40, LDH1 = 136, LDH2 = 72;     // padded row lengths (halfs)
  extern __shared__ char smem[];
  _Float16* inA = (_Float16*)smem;               // [POS][40]
  _Float16* h1  = inA + POS * LDI;               // [POS][136]
  _Float16* h2  = h1 + POS * LDH1;               // [POS][72]
  _Float16* w1  = h2 + POS * LDH2;               // [128][40]
  _Float16* w2  = w1 + 128 * LDI;                // [64][136]

  int blk = blockIdx.x;
  int b = blk / (N / PTS);
  int nbase = (blk % (N / PTS)) * PTS;
  int t = threadIdx.x;
  __builtin_prefetch(w1g, 0, 3);
  __builtin_prefetch(w2g, 0, 3);

  for (int i = t; i < 128 * LDI / 8; i += 256) ((uint4*)w1)[i] = ((const uint4*)w1g)[i];
  for (int i = t; i < 64 * LDH1 / 8; i += 256) ((uint4*)w2)[i] = ((const uint4*)w2g)[i];

  { // build one position-major row per thread: [c0..2 center, c3..5 diff, 0...]
    int p = t, pt = p >> 5, k = p & 31;
    int n = nbase + pt;
    int m = idx1[((size_t)b * N + n) * 32 + k];
    const float* xb = x + (size_t)b * 3 * N;
    Pack40 row;
#pragma unroll
    for (int c = 0; c < 40; ++c) row.h[c] = (_Float16)0.f;
#pragma unroll
    for (int c = 0; c < 3; ++c) {
      float cen = xb[c * N + n];
      float nbv = xb[c * N + m];
      row.h[c]     = (_Float16)cen;
      row.h[c + 3] = (_Float16)(nbv - cen);
    }
    uint4* dst = (uint4*)(inA + p * LDI);
#pragma unroll
    for (int u = 0; u < 5; ++u) dst[u] = row.u[u];
  }
  __syncthreads();

  int wave = t >> 5;
  // W1: 8 cout-tiles x 16 pos-tiles, single K=32 WMMA each
  for (int tile = wave; tile < 128; tile += 8) {
    int ct = tile >> 4, pti = tile & 15;
    v8f acc = {};
    v16h af = frag_a_rm(w1, LDI, ct * 16, 0);
    v16h bf = frag_b_pm(inA, LDI, 0, pti * 16);
    WMMA_F16(acc, af, bf);
    store_d_bn_lrelu(h1, LDH1, pti * 16, ct * 16, acc, s1, b1);
  }
  __syncthreads();
  // W2: 4 x 16 tiles, K loop over 128
  for (int tile = wave; tile < 64; tile += 8) {
    int ct = tile >> 4, pti = tile & 15;
    v8f acc = {};
#pragma unroll
    for (int kk = 0; kk < 128; kk += 32) {
      v16h af = frag_a_rm(w2, LDH1, ct * 16, kk);
      v16h bf = frag_b_pm(h1, LDH1, kk, pti * 16);
      WMMA_F16(acc, af, bf);
    }
    store_d_bn_lrelu(h2, LDH2, pti * 16, ct * 16, acc, s2, b2);
  }
  __syncthreads();
  // max over K -> f1 position-major (b,n,64)
  for (int i = t; i < 64 * PTS; i += 256) {
    int c = i >> 3, pt = i & 7;
    float mx = -3.4e38f;
#pragma unroll
    for (int k = 0; k < 32; ++k) mx = fmaxf(mx, (float)h2[(pt * 32 + k) * LDH2 + c]);
    size_t o = ((size_t)b * N + nbase + pt) * 64 + c;
    f1fT[o] = mx;
    f1hT[o] = (_Float16)mx;
  }
}

// ---------------------------------------------------------------------------
// 4) KNN in 64-D: gram tiles via WMMA, fused top-32 merge.
//    128 queries per 4-wave workgroup; neg-dist tile stored transposed.
// ---------------------------------------------------------------------------
__global__ __launch_bounds__(128) void k_knn2(const float* f1fT, const _Float16* f1hT,
                                              int* idx2, int B, int N) {
  const int LDF = 72, LDN = 132;
  alignas(16) __shared__ _Float16 Fq[128 * LDF];
  alignas(16) __shared__ _Float16 Fc[16 * LDF];
  alignas(16) __shared__ float    nd[16 * LDN];   // nd[cand][query]
  __shared__ float sqq[128];
  __shared__ float sqc[16];

  int blk = blockIdx.x;
  int b = blk / (N / 128);
  int qbase = (blk % (N / 128)) * 128;
  int t = threadIdx.x;
  int wave = t >> 5, lane = t & 31;

  { // query tile: contiguous uint4 copies, position-major
    const uint4* src = (const uint4*)(f1hT + ((size_t)b * N + qbase + t) * 64);
    uint4* dst = (uint4*)(Fq + t * LDF);
#pragma unroll
    for (int u = 0; u < 8; ++u) dst[u] = src[u];
    const float* fr = f1fT + ((size_t)b * N + qbase + t) * 64;
    float s = 0.f;
#pragma unroll
    for (int c = 0; c < 64; ++c) { float v = fr[c]; s += v * v; }
    sqq[t] = s;
  }
  float bv[32]; int bi[32];
#pragma unroll
  for (int k = 0; k < 32; ++k) { bv[k] = -3.4e38f; bi[k] = 0x7fffffff; }

  for (int ct = 0; ct < N / 16; ++ct) {
    if (t < 16) {
      const uint4* src = (const uint4*)(f1hT + ((size_t)b * N + ct * 16 + t) * 64);
      uint4* dst = (uint4*)(Fc + t * LDF);
#pragma unroll
      for (int u = 0; u < 8; ++u) dst[u] = src[u];
      const float* fr = f1fT + ((size_t)b * N + ct * 16 + t) * 64;
      float s = 0.f;
#pragma unroll
      for (int c = 0; c < 64; ++c) { float v = fr[c]; s += v * v; }
      sqc[t] = s;
    }
    __syncthreads();
    for (int qt = wave; qt < 8; qt += 4) {
      v8f acc = {};
#pragma unroll
      for (int kk = 0; kk < 64; kk += 32) {
        v16h af = frag_a_rm(Fq, LDF, qt * 16, kk);   // A[q][d]
        v16h bf = frag_b_pm(Fc, LDF, kk, 0);         // B[d][cand]
        WMMA_F16(acc, af, bf);
      }
      int n0 = lane & 15;                            // candidate
      int mb = qt * 16 + ((lane >> 4) << 3);         // query base
      float sc = sqc[n0];
      FragF fr; fr.v = acc;
      PackF4 a0, a1;
#pragma unroll
      for (int r = 0; r < 4; ++r) a0.f[r] = 2.f * fr.f[r] - sqq[mb + r] - sc;
#pragma unroll
      for (int r = 0; r < 4; ++r) a1.f[r] = 2.f * fr.f[r + 4] - sqq[mb + 4 + r] - sc;
      *(float4*)(nd + n0 * LDN + mb)     = a0.v;
      *(float4*)(nd + n0 * LDN + mb + 4) = a1.v;
    }
    __syncthreads();
    for (int j = 0; j < 16; ++j) {
      int m = ct * 16 + j;
      float v = nd[j * LDN + t];
      if (v > bv[31] || (v == bv[31] && m < bi[31])) {
        int jj = 31;
        while (jj > 0 && (bv[jj - 1] < v || (bv[jj - 1] == v && bi[jj - 1] > m))) {
          bv[jj] = bv[jj - 1]; bi[jj] = bi[jj - 1]; --jj;
        }
        bv[jj] = v; bi[jj] = m;
      }
    }
    __syncthreads();
  }
  for (int k = 0; k < 32; ++k) idx2[((size_t)b * N + qbase + t) * 32 + k] = bi[k];
}

// ---------------------------------------------------------------------------
// 5) stage2: gather 64-D [center,diff] (128ch) -> W3 -> W4 -> max over K
//    4 points (128 positions) per 8-wave workgroup.
// ---------------------------------------------------------------------------
__global__ __launch_bounds__(256) void k_stage2(
    const _Float16* f1hT, const int* idx2,
    const _Float16* w3g, const _Float16* w4g,
    const float* s3, const float* b3, const float* s4, const float* b4,
    _Float16* f2hT, int B, int N)
{
  const int POS = 128, PTS = 4;
  const int LDI = 136, LDH = 136, LDO = 72;
  extern __shared__ char smem[];
  _Float16* inA = (_Float16*)smem;               // [POS][136]
  _Float16* h3  = inA + POS * LDI;               // [POS][136]
  _Float16* h4  = h3 + POS * LDH;                // [POS][72]
  _Float16* w3  = h4 + POS * LDO;                // [128][136]
  _Float16* w4  = w3 + 128 * LDH;                // [64][136]

  int blk = blockIdx.x;
  int b = blk / (N / PTS);
  int nbase = (blk % (N / PTS)) * PTS;
  int t = threadIdx.x;
  int wave = t >> 5;
  __builtin_prefetch(w3g, 0, 3);
  __builtin_prefetch(w4g, 0, 3);

  for (int i = t; i < 128 * LDH / 8; i += 256) ((uint4*)w3)[i] = ((const uint4*)w3g)[i];
  for (int i = t; i < 64 * LDH / 8; i += 256)  ((uint4*)w4)[i] = ((const uint4*)w4g)[i];

  { // two threads per position: half 0 = center (copy), half 1 = diff
    int p = t >> 1, halfrow = t & 1;
    int pt = p >> 5, k = p & 31;
    int n = nbase + pt;
    const _Float16* cen = f1hT + ((size_t)b * N + n) * 64;
    if (halfrow == 0) {
      const uint4* src = (const uint4*)cen;
      uint4* dst = (uint4*)(inA + p * LDI);
#pragma unroll
      for (int u = 0; u < 8; ++u) dst[u] = src[u];
    } else {
      int m = idx2[((size_t)b * N + n) * 32 + k];
      const _Float16* nbr = f1hT + ((size_t)b * N + m) * 64;
      Pack64 row;
#pragma unroll
      for (int c = 0; c < 64; ++c)
        row.h[c] = (_Float16)((float)nbr[c] - (float)cen[c]);
      uint4* dst = (uint4*)(inA + p * LDI + 64);
#pragma unroll
      for (int u = 0; u < 8; ++u) dst[u] = row.u[u];
    }
  }
  __syncthreads();

  // W3: 8 cout-tiles x 8 pos-tiles, K loop over 128
  for (int tile = wave; tile < 64; tile += 8) {
    int ct = tile >> 3, pti = tile & 7;
    v8f acc = {};
#pragma unroll
    for (int kk = 0; kk < 128; kk += 32) {
      v16h af = frag_a_rm(w3, LDH, ct * 16, kk);
      v16h bf = frag_b_pm(inA, LDI, kk, pti * 16);
      WMMA_F16(acc, af, bf);
    }
    store_d_bn_lrelu(h3, LDH, pti * 16, ct * 16, acc, s3, b3);
  }
  __syncthreads();
  // W4: 4 x 8 tiles
  for (int tile = wave; tile < 32; tile += 8) {
    int ct = tile >> 3, pti = tile & 7;
    v8f acc = {};
#pragma unroll
    for (int kk = 0; kk < 128; kk += 32) {
      v16h af = frag_a_rm(w4, LDH, ct * 16, kk);
      v16h bf = frag_b_pm(h3, LDH, kk, pti * 16);
      WMMA_F16(acc, af, bf);
    }
    store_d_bn_lrelu(h4, LDO, pti * 16, ct * 16, acc, s4, b4);
  }
  __syncthreads();
  { // max over K: 64ch x 4pts = 256 -> one per thread
    int c = t >> 2, pt = t & 3;
    float mx = -3.4e38f;
#pragma unroll
    for (int k = 0; k < 32; ++k) mx = fmaxf(mx, (float)h4[(pt * 32 + k) * LDO + c]);
    f2hT[((size_t)b * N + nbase + pt) * 64 + c] = (_Float16)mx;
  }
}

// ---------------------------------------------------------------------------
// 6) stage3: concat(f1,f2) -> W5 + sin/cos positional encoding -> W6 -> out.
//    16 points per 4-wave workgroup; weight fragments gathered from global.
// ---------------------------------------------------------------------------
__global__ __launch_bounds__(128) void k_stage3(
    const float* x, const _Float16* f1hT, const _Float16* f2hT,
    const _Float16* w5g, const _Float16* w6g,
    const float* s5, const float* b5, const float* s6, const float* b6,
    const float* mnmx, float* out, int B, int N)
{
  const int LDF = 136, LDY = 392;
  alignas(16) __shared__ _Float16 fin[16 * LDF];
  alignas(16) __shared__ _Float16 y[16 * LDY];
  int blk = blockIdx.x;
  int b = blk / (N / 16);
  int nbase = (blk % (N / 16)) * 16;
  int t = threadIdx.x;
  int wave = t >> 5, lane = t & 31;
  __builtin_prefetch(w5g, 0, 3);
  __builtin_prefetch(w6g, 0, 3);

  { // fill fin position-major: 8 segments of 16 channels per point
    int j = t >> 3, part = t & 7;
    int n = nbase + j;
    const _Float16* src16 = (part < 4)
        ? f1hT + ((size_t)b * N + n) * 64 + part * 16
        : f2hT + ((size_t)b * N + n) * 64 + (part - 4) * 16;
    uint4* dst = (uint4*)(fin + j * LDF + part * 16);
    dst[0] = ((const uint4*)src16)[0];
    dst[1] = ((const uint4*)src16)[1];
  }
  __syncthreads();

  float mn = mnmx[0], mx = mnmx[1];
  float inv = 2.f / (mx - mn);

  // W5 (384x128) + positional encoding, written f16 position-major into y
  for (int ctile = wave; ctile < 24; ctile += 4) {
    v8f acc = {};
#pragma unroll
    for (int kk = 0; kk < 128; kk += 32) {
      v16h af = frag_a_rm(w5g, LDF, ctile * 16, kk);
      v16h bf = frag_b_pm(fin, LDF, kk, 0);
      WMMA_F16(acc, af, bf);
    }
    int j  = lane & 15;
    int mb = ctile * 16 + ((lane >> 4) << 3);
    int n  = nbase + j;
    FragF fr; fr.v = acc;
    Pack8 pk;
#pragma unroll
    for (int r = 0; r < 8; ++r) {
      int oc = mb + r;
      float v = lrelu(fr.f[r] * s5[oc] + b5[oc]);
      int c3 = oc >> 7;          // coordinate 0..2
      int tt = oc & 127;         // 0..63 sin, 64..127 cos
      float coord = x[((size_t)b * 3 + c3) * N + n];
      float normal = (coord - mn) * inv - 1.f;
      float ang = normal * 3.14159265358979323846f * exp2f((float)(tt & 63));
      v += (tt < 64) ? sinf(ang) : cosf(ang);
      pk.h[r] = (_Float16)v;
    }
    *(uint4*)(y + j * LDY + mb) = pk.u;
  }
  __syncthreads();
  // W6 (64x384)
  for (int ctile = wave; ctile < 4; ctile += 4) {
    v8f acc = {};
#pragma unroll
    for (int kk = 0; kk < 384; kk += 32) {
      v16h af = frag_a_rm(w6g, LDY, ctile * 16, kk);
      v16h bf = frag_b_pm(y, LDY, kk, 0);
      WMMA_F16(acc, af, bf);
    }
    int j  = lane & 15;
    int mb = ctile * 16 + ((lane >> 4) << 3);
    int n  = nbase + j;
    FragF fr; fr.v = acc;
#pragma unroll
    for (int r = 0; r < 8; ++r) {
      int oc = mb + r;
      out[((size_t)b * 64 + oc) * N + n] = lrelu(fr.f[r] * s6[oc] + b6[oc]);
    }
  }
}

// ---------------------------------------------------------------------------
extern "C" void kernel_launch(void* const* d_in, const int* in_sizes, int n_in,
                              void* d_out, int out_size, void* d_ws, size_t ws_size,
                              hipStream_t stream) {
  const float* x  = (const float*)d_in[0];
  const float* W1 = (const float*)d_in[1];  const float* s1 = (const float*)d_in[2];  const float* b1 = (const float*)d_in[3];
  const float* W2 = (const float*)d_in[4];  const float* s2 = (const float*)d_in[5];  const float* b2 = (const float*)d_in[6];
  const float* W3 = (const float*)d_in[7];  const float* s3 = (const float*)d_in[8];  const float* b3 = (const float*)d_in[9];
  const float* W4 = (const float*)d_in[10]; const float* s4 = (const float*)d_in[11]; const float* b4 = (const float*)d_in[12];
  const float* W5 = (const float*)d_in[13]; const float* s5 = (const float*)d_in[14]; const float* b5 = (const float*)d_in[15];
  const float* W6 = (const float*)d_in[16]; const float* s6 = (const float*)d_in[17]; const float* b6 = (const float*)d_in[18];
  float* out = (float*)d_out;

  const int B = 2;
  const int N = in_sizes[0] / (3 * B);   // 4096

  char* ws = (char*)d_ws;
  size_t off = 0;
  auto carve = [&](size_t bytes) -> char* {
    char* p = ws + off;
    off += (bytes + 255) & ~(size_t)255;
    return p;
  };
  _Float16* w1h  = (_Float16*)carve((size_t)128 * 40 * 2);
  _Float16* w2h  = (_Float16*)carve((size_t)64 * 136 * 2);
  _Float16* w3h  = (_Float16*)carve((size_t)128 * 136 * 2);
  _Float16* w4h  = (_Float16*)carve((size_t)64 * 136 * 2);
  _Float16* w5h  = (_Float16*)carve((size_t)384 * 136 * 2);
  _Float16* w6h  = (_Float16*)carve((size_t)64 * 392 * 2);
  float*    mnmx = (float*)carve(2 * sizeof(float));
  int*      idx1 = (int*)carve((size_t)B * N * 32 * 4);
  int*      idx2 = (int*)carve((size_t)B * N * 32 * 4);
  float*    f1fT = (float*)carve((size_t)B * N * 64 * 4);
  _Float16* f1hT = (_Float16*)carve((size_t)B * N * 64 * 2);
  _Float16* f2hT = (_Float16*)carve((size_t)B * N * 64 * 2);
  (void)ws_size; (void)n_in; (void)out_size;

  k_minmax<<<1, 256, 0, stream>>>(x, B * 3 * N, mnmx);
  k_cvt_pad<<<(128 * 40  + 255) / 256, 256, 0, stream>>>(W1, w1h, 128, 6,   40);
  k_cvt_pad<<<(64  * 136 + 255) / 256, 256, 0, stream>>>(W2, w2h, 64,  128, 136);
  k_cvt_pad<<<(128 * 136 + 255) / 256, 256, 0, stream>>>(W3, w3h, 128, 128, 136);
  k_cvt_pad<<<(64  * 136 + 255) / 256, 256, 0, stream>>>(W4, w4h, 64,  128, 136);
  k_cvt_pad<<<(384 * 136 + 255) / 256, 256, 0, stream>>>(W5, w5h, 384, 128, 136);
  k_cvt_pad<<<(64  * 392 + 255) / 256, 256, 0, stream>>>(W6, w6h, 64,  384, 392);

  k_knn1<<<B * (N / 256), 256, (size_t)4 * N * 4, stream>>>(x, idx1, B, N);

  size_t sh1 = ((size_t)256 * 40 + 256 * 136 + 256 * 72 + 128 * 40 + 64 * 136) * 2;  // ~151 KB
  k_stage1<<<B * (N / 8), 256, sh1, stream>>>(x, idx1, w1h, w2h,
                                              s1, b1, s2, b2, f1fT, f1hT, B, N);

  k_knn2<<<B * (N / 128), 128, 0, stream>>>(f1fT, f1hT, idx2, B, N);

  size_t sh2 = ((size_t)128 * 136 + 128 * 136 + 128 * 72 + 128 * 136 + 64 * 136) * 2; // ~137 KB
  k_stage2<<<B * (N / 4), 256, sh2, stream>>>(f1hT, idx2, w3h, w4h,
                                              s3, b3, s4, b4, f2hT, B, N);

  k_stage3<<<B * (N / 16), 128, 0, stream>>>(x, f1hT, f2hT, w5h, w6h,
                                             s5, b5, s6, b6, mnmx, out, B, N);
}
